// get_model_34514357191290
// MI455X (gfx1250) — matrix operand order repaired and checked by reading, not compiled, not run on hardware
//
#include <hip/hip_runtime.h>
#include <hip/hip_bf16.h>

// ---------------------------------------------------------------------------
// PointNet++ MSG classifier forward for MI455X (gfx1250, wave32, WMMA).
// All dense math (26 conv-MLP layers + 6 FC layers) runs through one fused
// WMMA f16 GEMM + bias + BN + ReLU kernel (v_wmma_f32_16x16x32_f16, f32 acc).
// Weights are pre-packed once per layer into zero-padded f16 (k_prep_w), so
// the GEMM inner loop is pure vector loads + WMMA (no guards / converts).
// Activations live in ws as f16 with channel counts padded to multiples of 32
// (pad lanes written as zeros by every producer kernel -> no memsets needed).
// ---------------------------------------------------------------------------

typedef _Float16 f16;
typedef _Float16 v16h __attribute__((ext_vector_type(16)));
typedef _Float16 h8   __attribute__((ext_vector_type(8)));
typedef float    v8f  __attribute__((ext_vector_type(8)));

// ---------------- transpose [B,3,N] -> [B,N,3] ----------------
__global__ void k_transpose_xyz(const float* __restrict__ in, float* __restrict__ out,
                                int N, int total) {
  int i = blockIdx.x * blockDim.x + threadIdx.x;
  if (i >= total) return;
  int b = i / N, n = i % N;
  out[(size_t)i * 3 + 0] = in[((size_t)b * 3 + 0) * N + n];
  out[(size_t)i * 3 + 1] = in[((size_t)b * 3 + 1) * N + n];
  out[(size_t)i * 3 + 2] = in[((size_t)b * 3 + 2) * N + n];
}

// ---------------- farthest point sampling: 1 block (1024 thr) per batch ----
__global__ __launch_bounds__(1024) void k_fps(const float* __restrict__ xyz, int N,
                                              int npoint, int* __restrict__ out) {
  int b = blockIdx.x;
  const float* p = xyz + (size_t)b * N * 3;
  const int t = threadIdx.x;
  const int lane = t & 31, wv = t >> 5;
  float dist[16];                      // N <= 16384, 1024 threads -> <=16 pts/thr
  int ppt = (N + 1023) >> 10;
  for (int j = 0; j < ppt; ++j) dist[j] = 1e10f;
  __shared__ float s_val[32];
  __shared__ int   s_idx[32];
  __shared__ int   s_far;
  if (t == 0) s_far = 0;
  __syncthreads();
  for (int it = 0; it < npoint; ++it) {
    int far = s_far;
    if (t == 0) out[b * npoint + it] = far;
    float cx = p[far * 3 + 0], cy = p[far * 3 + 1], cz = p[far * 3 + 2];
    float best = -1.0f; int bi = 0;
    for (int j = 0; j < ppt; ++j) {
      int i = t + (j << 10);
      if (i < N) {
        float dx = p[i * 3 + 0] - cx, dy = p[i * 3 + 1] - cy, dz = p[i * 3 + 2] - cz;
        float d = dx * dx + dy * dy + dz * dz;
        float m = fminf(dist[j], d);
        dist[j] = m;
        if (m > best) { best = m; bi = i; }
      }
    }
#pragma unroll
    for (int off = 16; off > 0; off >>= 1) {     // wave32 reduction
      float v2 = __shfl_down(best, off);
      int   i2 = __shfl_down(bi, off);
      if (v2 > best) { best = v2; bi = i2; }
    }
    if (lane == 0) { s_val[wv] = best; s_idx[wv] = bi; }
    __syncthreads();
    if (t < 32) {
      best = s_val[t]; bi = s_idx[t];
#pragma unroll
      for (int off = 16; off > 0; off >>= 1) {
        float v2 = __shfl_down(best, off);
        int   i2 = __shfl_down(bi, off);
        if (v2 > best) { best = v2; bi = i2; }
      }
      if (t == 0) s_far = bi;
    }
    __syncthreads();
  }
}

// ---------------- gather centroid xyz ----------------
__global__ void k_gather_xyz(const float* __restrict__ xyz, const int* __restrict__ idx,
                             float* __restrict__ out, int N, int M, int total) {
  int i = blockIdx.x * blockDim.x + threadIdx.x;
  if (i >= total) return;
  int b = i / M;
  int j = idx[i];
  const float* p = xyz + ((size_t)b * N + j) * 3;
  out[(size_t)i * 3 + 0] = p[0];
  out[(size_t)i * 3 + 1] = p[1];
  out[(size_t)i * 3 + 2] = p[2];
}

// ---------------- ball query: first-K indices within radius (sorted order) --
__global__ void k_ball_query(const float* __restrict__ xyz, int N,
                             const float* __restrict__ cen, int M,
                             float r2, int K, int* __restrict__ out, int total) {
  int i = blockIdx.x * blockDim.x + threadIdx.x;
  if (i >= total) return;                  // total = B*M
  int b = i / M;
  const float* p = xyz + (size_t)b * N * 3;
  const float* c = cen + (size_t)i * 3;
  float cx = c[0], cy = c[1], cz = c[2];
  int* o = out + (size_t)i * K;
  int cnt = 0, first = -1;
  for (int j = 0; j < N && cnt < K; ++j) {
    float dx = p[j * 3 + 0] - cx, dy = p[j * 3 + 1] - cy, dz = p[j * 3 + 2] - cz;
    float d = dx * dx + dy * dy + dz * dz;
    if (d <= r2) { if (first < 0) first = j; o[cnt++] = j; }
  }
  if (first < 0) first = 0;
  while (cnt < K) o[cnt++] = first;
}

// ---------------- build grouped features (rel-xyz + gathered points), f16 ---
__global__ void k_group_feat(const float* __restrict__ xyz, const float* __restrict__ cen,
                             const int* __restrict__ idx, const f16* __restrict__ pts,
                             int Cp, int ldp, int N, int M, int K, int xyz_first,
                             f16* __restrict__ feat, int ldf, int total) {
  int i = blockIdx.x * blockDim.x + threadIdx.x;
  if (i >= total) return;                  // total = B*M*K
  int bm = i / K;
  int b  = bm / M;
  int j  = idx[i];
  const float* p = xyz + ((size_t)b * N + j) * 3;
  const float* c = cen + (size_t)bm * 3;
  float gx = p[0] - c[0], gy = p[1] - c[1], gz = p[2] - c[2];
  f16* f = feat + (size_t)i * ldf;
  const f16* q = pts ? (pts + ((size_t)b * N + j) * ldp) : (const f16*)0;
  int pos = 0;
  if (xyz_first) { f[pos++] = (f16)gx; f[pos++] = (f16)gy; f[pos++] = (f16)gz; }
  for (int c2 = 0; c2 < Cp; ++c2) f[pos++] = q[c2];
  if (!xyz_first) { f[pos++] = (f16)gx; f[pos++] = (f16)gy; f[pos++] = (f16)gz; }
  for (; pos < ldf; ++pos) f[pos] = (f16)0.f;   // zero the K-pad lanes
}

// ---------------- group_all feature for pn3: concat(xyz, pts) ----------------
__global__ void k_groupall(const float* __restrict__ xyz, const f16* __restrict__ pts,
                           int C, int ldp, f16* __restrict__ feat, int ldf, int total) {
  int i = blockIdx.x * blockDim.x + threadIdx.x;
  if (i >= total) return;
  const float* p = xyz + (size_t)i * 3;
  const f16* q = pts + (size_t)i * ldp;
  f16* f = feat + (size_t)i * ldf;
  int pos = 0;
  f[pos++] = (f16)p[0]; f[pos++] = (f16)p[1]; f[pos++] = (f16)p[2];
  for (int c = 0; c < C; ++c) f[pos++] = q[c];
  for (; pos < ldf; ++pos) f[pos] = (f16)0.f;
}

// ---------------- 3-NN inverse-distance interpolation (+optional concat) ----
__global__ void k_fp_interp(const float* __restrict__ xyz1, int N1,
                            const float* __restrict__ xyz2, int S,
                            const f16* __restrict__ pts2, int C2, int ld2,
                            const f16* __restrict__ pts1, int C1, int ld1,
                            f16* __restrict__ out, int ldo, int total) {
  int i = blockIdx.x * blockDim.x + threadIdx.x;
  if (i >= total) return;                  // total = B*N1
  int b = i / N1;
  const float* p1 = xyz1 + (size_t)i * 3;
  const float* p2 = xyz2 + (size_t)b * S * 3;
  float d0 = 1e30f, d1 = 1e30f, d2 = 1e30f;
  int   i0 = 0, i1 = 0, i2 = 0;
  float ax = p1[0], ay = p1[1], az = p1[2];
  for (int s = 0; s < S; ++s) {
    float dx = p2[s * 3 + 0] - ax, dy = p2[s * 3 + 1] - ay, dz = p2[s * 3 + 2] - az;
    float d = dx * dx + dy * dy + dz * dz;
    if (d < d0)      { d2 = d1; i2 = i1; d1 = d0; i1 = i0; d0 = d; i0 = s; }
    else if (d < d1) { d2 = d1; i2 = i1; d1 = d;  i1 = s; }
    else if (d < d2) { d2 = d;  i2 = s; }
  }
  float w0 = 1.f / (d0 + 1e-8f), w1 = 1.f / (d1 + 1e-8f), w2 = 1.f / (d2 + 1e-8f);
  float ws = w0 + w1 + w2;
  w0 /= ws; w1 /= ws; w2 /= ws;
  const f16* q = pts2 + (size_t)b * S * ld2;
  f16* o = out + (size_t)i * ldo;
  int pos = 0;
  for (int c = 0; c < C1; ++c) o[pos++] = pts1[(size_t)i * ld1 + c];  // points1 first
  for (int c = 0; c < C2; ++c) {
    float v = w0 * (float)q[(size_t)i0 * ld2 + c]
            + w1 * (float)q[(size_t)i1 * ld2 + c]
            + w2 * (float)q[(size_t)i2 * ld2 + c];
    o[pos++] = (f16)v;
  }
  for (; pos < ldo; ++pos) o[pos] = (f16)0.f;
}

// ---------------- weight pre-pack: f32 [Cout,Cin] -> f16 [ldy,ldx] zero-pad --
__global__ void k_prep_w(const float* __restrict__ W, int Cin, int Cout,
                         f16* __restrict__ Wp, int ldx, int total) {
  int i = blockIdx.x * blockDim.x + threadIdx.x;
  if (i >= total) return;                  // total = ldy * ldx
  int d = i / ldx, k = i % ldx;
  float v = (d < Cout && k < Cin) ? W[(size_t)d * Cin + k] : 0.f;
  Wp[i] = (f16)v;
}

// ---------------- WMMA GEMM: Y = act(BN((X * Wp^T) + b)), f16 in/out, f32 acc
// X:  [M, ldx] f16 (ldx % 32 == 0, pad cols zero)
// Wp: [ldy, ldx] f16 pre-packed, zero-padded   Y: [M, ldy] f16 (pads zeroed)
// 256 threads = 8 wave32; wave w -> output tile (blockIdx.x, blockIdx.y*8+w).
// Inner loop: 2x b128 A loads + 32B B load + v_wmma, K unrolled x2 with dual
// accumulators to break the WMMA->WMMA RAW chain.
__global__ __launch_bounds__(256) void k_gemm(
    const f16* __restrict__ X, int ldx, const f16* __restrict__ Wp,
    const float* __restrict__ bias, const float* __restrict__ g,
    const float* __restrict__ be, int Cout, f16* __restrict__ Y,
    int ldy, int relu) {
  const int wave = threadIdx.x >> 5;
  const int lane = threadIdx.x & 31;
  const int r  = lane & 15;
  const int hi = lane >> 4;
  const int ntile = blockIdx.y * 8 + wave;
  if (ntile * 16 >= ldy) return;           // wave-uniform exit: EXEC stays full
  const int m0 = blockIdx.x * 16;
  const int d  = ntile * 16 + r;           // output channel owned by this lane

  v8f acc0 = {0.f, 0.f, 0.f, 0.f, 0.f, 0.f, 0.f, 0.f};
  v8f acc1 = {0.f, 0.f, 0.f, 0.f, 0.f, 0.f, 0.f, 0.f};
  const f16* xrow = X  + (size_t)(m0 + r) * ldx;  // A row m0+r (both half-waves)
  const f16* brow = Wp + (size_t)d * ldx;         // B row = channel d
  const int kt = ldx >> 5;

  int kk = 0;
  for (; kk + 1 < kt; kk += 2) {
    const int k0 = kk << 5;
    __builtin_prefetch(xrow + k0 + 64, 0, 0);     // global_prefetch_b8 on A stream
    // --- chunk 0 ---
    h8 alo0 = *(const h8*)(xrow + k0 + 16 * hi);
    h8 ahi0 = *(const h8*)(xrow + k0 + 8 + 16 * hi);
    v16h Bf0 = *(const v16h*)(brow + k0 + 16 * hi);
    v16h Af0;
#pragma unroll
    for (int j = 0; j < 8; ++j) { Af0[j] = alo0[j]; Af0[j + 8] = ahi0[j]; }
    // --- chunk 1 ---
    h8 alo1 = *(const h8*)(xrow + k0 + 32 + 16 * hi);
    h8 ahi1 = *(const h8*)(xrow + k0 + 40 + 16 * hi);
    v16h Bf1 = *(const v16h*)(brow + k0 + 32 + 16 * hi);
    v16h Af1;
#pragma unroll
    for (int j = 0; j < 8; ++j) { Af1[j] = alo1[j]; Af1[j + 8] = ahi1[j]; }
    acc0 = __builtin_amdgcn_wmma_f32_16x16x32_f16(false, Af0, false, Bf0,
                                                  (short)0, acc0, false, false);
    acc1 = __builtin_amdgcn_wmma_f32_16x16x32_f16(false, Af1, false, Bf1,
                                                  (short)0, acc1, false, false);
  }
  if (kk < kt) {                                   // odd-kt tail
    const int k0 = kk << 5;
    h8 alo0 = *(const h8*)(xrow + k0 + 16 * hi);
    h8 ahi0 = *(const h8*)(xrow + k0 + 8 + 16 * hi);
    v16h Bf0 = *(const v16h*)(brow + k0 + 16 * hi);
    v16h Af0;
#pragma unroll
    for (int j = 0; j < 8; ++j) { Af0[j] = alo0[j]; Af0[j + 8] = ahi0[j]; }
    acc0 = __builtin_amdgcn_wmma_f32_16x16x32_f16(false, Af0, false, Bf0,
                                                  (short)0, acc0, false, false);
  }
#pragma unroll
  for (int j = 0; j < 8; ++j) acc0[j] += acc1[j];

  // Epilogue. C/D layout: VGPR i -> M = i + 8*hi, N = lane&15 = channel d.
  // Branch-free: clamp channel index for param loads, select-zero pad lanes.
  const int  dcl  = (d < Cout) ? d : (Cout - 1);
  const bool live = (d < Cout);
  float bb = bias ? bias[dcl] : 0.f;
  float scv = 1.f, shv = 0.f;
  if (g) { scv = g[dcl] * rsqrtf(1.f + 1e-5f); shv = be[dcl]; }
#pragma unroll
  for (int i2 = 0; i2 < 8; ++i2) {
    int m = m0 + i2 + 8 * hi;
    float v = (acc0[i2] + bb) * scv + shv;
    if (relu) v = fmaxf(v, 0.f);
    v = live ? v : 0.f;                    // pads (d >= Cout) written as zero
    Y[(size_t)m * ldy + d] = (f16)v;
  }
}

// ---------------- max over K grouped rows ----------------
__global__ void k_maxpool(const f16* __restrict__ X, int ld, int K,
                          f16* __restrict__ Y, int total) {
  int i = blockIdx.x * blockDim.x + threadIdx.x;
  if (i >= total) return;                  // total = rows_out * ld
  int row = i / ld, c = i % ld;
  const f16* x = X + (size_t)row * K * ld + c;
  float m = (float)x[0];
  for (int k = 1; k < K; ++k) m = fmaxf(m, (float)x[(size_t)k * ld]);
  Y[(size_t)row * ld + c] = (f16)m;
}

// ---------------- channel-block copy (concat) ----------------
__global__ void k_copy_ch(const f16* __restrict__ src, int lds_,
                          f16* __restrict__ dst, int ldd, int off, int C, int total) {
  int i = blockIdx.x * blockDim.x + threadIdx.x;
  if (i >= total) return;                  // total = rows * C
  int row = i / C, c = i % C;
  dst[(size_t)row * ldd + off + c] = src[(size_t)row * lds_ + c];
}

// ---------------- head: log_softmax(class) + copy reg, f32 out ----------------
__global__ void k_head(const f16* __restrict__ logits, int ldl, int ncls,
                       const f16* __restrict__ reg, int ldr, int nreg,
                       float* __restrict__ out, int B) {
  int b = blockIdx.x * blockDim.x + threadIdx.x;
  if (b >= B) return;
  float mx = -1e30f;
  for (int c = 0; c < ncls; ++c) mx = fmaxf(mx, (float)logits[b * ldl + c]);
  float s = 0.f;
  for (int c = 0; c < ncls; ++c) s += expf((float)logits[b * ldl + c] - mx);
  float ls = logf(s);
  for (int c = 0; c < ncls; ++c)
    out[b * ncls + c] = ((float)logits[b * ldl + c] - mx) - ls;
  for (int j = 0; j < nreg; ++j)
    out[B * ncls + b * nreg + j] = (float)reg[b * ldr + j];
}

// ---------------------------------------------------------------------------
struct LP { const float *W, *b, *g, *be; };

extern "C" void kernel_launch(void* const* d_in, const int* in_sizes, int n_in,
                              void* d_out, int out_size, void* d_ws, size_t ws_size,
                              hipStream_t stream) {
  (void)in_sizes; (void)n_in; (void)out_size; (void)ws_size;
  const int B = 16, N0 = 16384;
  const float* xyz_in = (const float*)d_in[0];

  // ---- parameter table (dict-order flatten: W,b,g,be per conv layer) ----
  int pi = 1;
  auto nextLP = [&]() {
    LP L{(const float*)d_in[pi], (const float*)d_in[pi + 1],
         (const float*)d_in[pi + 2], (const float*)d_in[pi + 3]};
    pi += 4; return L;
  };
  LP sa1b0[3], sa1b1[3], sa2b0[3], sa2b1[3], fp2L[2], fp1L[3], pn1L[3], pn2L[3], pn3L[3];
  for (int i = 0; i < 3; ++i) sa1b0[i] = nextLP();
  for (int i = 0; i < 3; ++i) sa1b1[i] = nextLP();
  for (int i = 0; i < 3; ++i) sa2b0[i] = nextLP();
  for (int i = 0; i < 3; ++i) sa2b1[i] = nextLP();
  for (int i = 0; i < 2; ++i) fp2L[i]  = nextLP();
  for (int i = 0; i < 3; ++i) fp1L[i]  = nextLP();
  for (int i = 0; i < 3; ++i) pn1L[i]  = nextLP();
  for (int i = 0; i < 3; ++i) pn2L[i]  = nextLP();
  for (int i = 0; i < 3; ++i) pn3L[i]  = nextLP();
  const float* fc1W = (const float*)d_in[pi++]; const float* fc1b = (const float*)d_in[pi++];
  const float* bn1g = (const float*)d_in[pi++]; const float* bn1e = (const float*)d_in[pi++];
  const float* fc2W = (const float*)d_in[pi++]; const float* fc2b = (const float*)d_in[pi++];
  const float* bn2g = (const float*)d_in[pi++]; const float* bn2e = (const float*)d_in[pi++];
  const float* fc3W = (const float*)d_in[pi++]; const float* fc3b = (const float*)d_in[pi++];
  const float* fc4W = (const float*)d_in[pi++]; const float* fc4b = (const float*)d_in[pi++];
  const float* fc5W = (const float*)d_in[pi++]; const float* fc5b = (const float*)d_in[pi++];
  const float* fc6W = (const float*)d_in[pi++]; const float* fc6b = (const float*)d_in[pi++];

  // ---- workspace bump allocator (deterministic) ----
  char* wp = (char*)d_ws;
  auto alloc = [&](size_t n) -> void* {
    void* r = (void*)wp; wp += (n + 255) & ~(size_t)255; return r;
  };
  auto cdiv = [](long a, long b) { return (unsigned)((a + b - 1) / b); };
  // pre-pack weights to f16 then run the WMMA GEMM
  auto gemm = [&](const void* X, int ldx, const LP& L, int cin, int cout,
                  void* Y, int ldy, long M, int relu) {
    f16* Wp = (f16*)alloc((size_t)ldy * ldx * 2);
    int wtot = ldy * ldx;
    k_prep_w<<<cdiv(wtot, 256), 256, 0, stream>>>(L.W, cin, cout, Wp, ldx, wtot);
    dim3 gd((unsigned)(M / 16), (unsigned)((ldy / 16 + 7) / 8));
    k_gemm<<<gd, 256, 0, stream>>>((const f16*)X, ldx, Wp, L.b, L.g, L.be,
                                   cout, (f16*)Y, ldy, relu);
  };

  // ---- l0_xyz = transpose(xyz) : [B,N0,3] ----
  float* l0xyz = (float*)alloc((size_t)B * N0 * 3 * 4);
  k_transpose_xyz<<<cdiv((long)B * N0, 256), 256, 0, stream>>>(xyz_in, l0xyz, N0, B * N0);

  // ================= SA1 (MSG, npoint=64) =================
  int* idx1 = (int*)alloc(B * 64 * 4);
  k_fps<<<B, 1024, 0, stream>>>(l0xyz, N0, 64, idx1);
  float* nx1 = (float*)alloc(B * 64 * 3 * 4);
  k_gather_xyz<<<cdiv(B * 64, 256), 256, 0, stream>>>(l0xyz, idx1, nx1, N0, 64, B * 64);

  // branch0: r=.05 K=16, MLP 3->16->16->32
  int* gi0 = (int*)alloc((size_t)B * 64 * 16 * 4);
  k_ball_query<<<cdiv(B * 64, 256), 256, 0, stream>>>(l0xyz, N0, nx1, 64, 0.0025f, 16, gi0, B * 64);
  long Ma = (long)B * 64 * 16;
  f16* fa = (f16*)alloc(Ma * 32 * 2);
  k_group_feat<<<cdiv(Ma, 256), 256, 0, stream>>>(l0xyz, nx1, gi0, (const f16*)0, 0, 0,
                                                  N0, 64, 16, 1, fa, 32, (int)Ma);
  f16* ta = (f16*)alloc(Ma * 32 * 2);
  f16* tb = (f16*)alloc(Ma * 32 * 2);
  gemm(fa, 32, sa1b0[0], 3, 16, ta, 32, Ma, 1);
  gemm(ta, 32, sa1b0[1], 16, 16, tb, 32, Ma, 1);
  gemm(tb, 32, sa1b0[2], 16, 32, ta, 32, Ma, 1);
  f16* pool0 = (f16*)alloc((size_t)B * 64 * 32 * 2);
  k_maxpool<<<cdiv(B * 64 * 32, 256), 256, 0, stream>>>(ta, 32, 16, pool0, B * 64 * 32);

  // branch1: r=.1 K=32, MLP 3->32->32->64
  int* gi1 = (int*)alloc((size_t)B * 64 * 32 * 4);
  k_ball_query<<<cdiv(B * 64, 256), 256, 0, stream>>>(l0xyz, N0, nx1, 64, 0.01f, 32, gi1, B * 64);
  long Mb = (long)B * 64 * 32;
  f16* fb = (f16*)alloc(Mb * 32 * 2);
  k_group_feat<<<cdiv(Mb, 256), 256, 0, stream>>>(l0xyz, nx1, gi1, (const f16*)0, 0, 0,
                                                  N0, 64, 32, 1, fb, 32, (int)Mb);
  f16* ua = (f16*)alloc(Mb * 64 * 2);
  f16* ub = (f16*)alloc(Mb * 64 * 2);
  gemm(fb, 32, sa1b1[0], 3, 32, ua, 32, Mb, 1);
  gemm(ua, 32, sa1b1[1], 32, 32, ub, 32, Mb, 1);
  gemm(ub, 32, sa1b1[2], 32, 64, ua, 64, Mb, 1);
  f16* pool1 = (f16*)alloc((size_t)B * 64 * 64 * 2);
  k_maxpool<<<cdiv(B * 64 * 64, 256), 256, 0, stream>>>(ua, 64, 32, pool1, B * 64 * 64);

  // l1_pts = concat -> [B*64, 96]
  f16* l1p = (f16*)alloc((size_t)B * 64 * 96 * 2);
  k_copy_ch<<<cdiv(B * 64 * 32, 256), 256, 0, stream>>>(pool0, 32, l1p, 96, 0, 32, B * 64 * 32);
  k_copy_ch<<<cdiv(B * 64 * 64, 256), 256, 0, stream>>>(pool1, 64, l1p, 96, 32, 64, B * 64 * 64);

  // ================= SA2 (MSG, npoint=16) =================
  int* idx2 = (int*)alloc(B * 16 * 4);
  k_fps<<<B, 1024, 0, stream>>>(nx1, 64, 16, idx2);
  float* nx2 = (float*)alloc(B * 16 * 3 * 4);
  k_gather_xyz<<<1, 256, 0, stream>>>(nx1, idx2, nx2, 64, 16, B * 16);

  // branch0: r=.1 K=16, feat 96+3=99 (pad 128), MLP 99->64->64->128
  int* g2a = (int*)alloc((size_t)B * 16 * 16 * 4);
  k_ball_query<<<1, 256, 0, stream>>>(nx1, 64, nx2, 16, 0.01f, 16, g2a, B * 16);
  long M2a = (long)B * 16 * 16;
  f16* f2a = (f16*)alloc(M2a * 128 * 2);
  k_group_feat<<<cdiv(M2a, 256), 256, 0, stream>>>(nx1, nx2, g2a, l1p, 96, 96,
                                                   64, 16, 16, 0, f2a, 128, (int)M2a);
  f16* wa = (f16*)alloc(M2a * 128 * 2);
  f16* wb = (f16*)alloc(M2a * 128 * 2);
  gemm(f2a, 128, sa2b0[0], 99, 64, wa, 64, M2a, 1);
  gemm(wa, 64, sa2b0[1], 64, 64, wb, 64, M2a, 1);
  gemm(wb, 64, sa2b0[2], 64, 128, wa, 128, M2a, 1);
  f16* pool2a = (f16*)alloc((size_t)B * 16 * 128 * 2);
  k_maxpool<<<cdiv(B * 16 * 128, 256), 256, 0, stream>>>(wa, 128, 16, pool2a, B * 16 * 128);

  // branch1: r=.2 K=32, MLP 99->64->96->128
  int* g2b = (int*)alloc((size_t)B * 16 * 32 * 4);
  k_ball_query<<<1, 256, 0, stream>>>(nx1, 64, nx2, 16, 0.04f, 32, g2b, B * 16);
  long M2b = (long)B * 16 * 32;
  f16* f2b = (f16*)alloc(M2b * 128 * 2);
  k_group_feat<<<cdiv(M2b, 256), 256, 0, stream>>>(nx1, nx2, g2b, l1p, 96, 96,
                                                   64, 16, 32, 0, f2b, 128, (int)M2b);
  f16* xa = (f16*)alloc(M2b * 128 * 2);
  f16* xb = (f16*)alloc(M2b * 128 * 2);
  gemm(f2b, 128, sa2b1[0], 99, 64, xa, 64, M2b, 1);
  gemm(xa, 64, sa2b1[1], 64, 96, xb, 96, M2b, 1);
  gemm(xb, 96, sa2b1[2], 96, 128, xa, 128, M2b, 1);
  f16* pool2b = (f16*)alloc((size_t)B * 16 * 128 * 2);
  k_maxpool<<<cdiv(B * 16 * 128, 256), 256, 0, stream>>>(xa, 128, 32, pool2b, B * 16 * 128);

  f16* l2p = (f16*)alloc((size_t)B * 16 * 256 * 2);
  k_copy_ch<<<cdiv(B * 16 * 128, 256), 256, 0, stream>>>(pool2a, 128, l2p, 256, 0, 128, B * 16 * 128);
  k_copy_ch<<<cdiv(B * 16 * 128, 256), 256, 0, stream>>>(pool2b, 128, l2p, 256, 128, 128, B * 16 * 128);

  // ================= FP2: interp l2 -> l1, concat l1p, MLP 352->256->128 ===
  long Mf2 = (long)B * 64;
  f16* ffp2 = (f16*)alloc(Mf2 * 352 * 2);
  k_fp_interp<<<cdiv(Mf2, 256), 256, 0, stream>>>(nx1, 64, nx2, 16, l2p, 256, 256,
                                                  l1p, 96, 96, ffp2, 352, (int)Mf2);
  f16* ya  = (f16*)alloc(Mf2 * 256 * 2);
  f16* l1n = (f16*)alloc(Mf2 * 128 * 2);
  gemm(ffp2, 352, fp2L[0], 352, 256, ya, 256, Mf2, 1);
  gemm(ya, 256, fp2L[1], 256, 128, l1n, 128, Mf2, 1);

  // ================= FP1: interp l1n -> all 16384 pts, MLP 128^3 (heavy) ===
  long Mf1 = (long)B * N0;                       // 262144 rows
  f16* bigA = (f16*)alloc((size_t)Mf1 * 128 * 2);
  f16* bigB = (f16*)alloc((size_t)Mf1 * 128 * 2);
  k_fp_interp<<<cdiv(Mf1, 256), 256, 0, stream>>>(l0xyz, N0, nx1, 64, l1n, 128, 128,
                                                  (const f16*)0, 0, 0, bigA, 128, (int)Mf1);
  gemm(bigA, 128, fp1L[0], 128, 128, bigB, 128, Mf1, 1);
  gemm(bigB, 128, fp1L[1], 128, 128, bigA, 128, Mf1, 1);
  gemm(bigA, 128, fp1L[2], 128, 128, bigB, 128, Mf1, 1);
  f16* l0p = bigB;                               // [B*N0, 128]

  // ================= pn1: SA(npoint=64, r=.2, K=64), feat 3+128=131 ========
  int* idxq = (int*)alloc(B * 64 * 4);
  k_fps<<<B, 1024, 0, stream>>>(l0xyz, N0, 64, idxq);
  float* nq1 = (float*)alloc(B * 64 * 3 * 4);
  k_gather_xyz<<<cdiv(B * 64, 256), 256, 0, stream>>>(l0xyz, idxq, nq1, N0, 64, B * 64);
  int* gq1 = (int*)alloc((size_t)B * 64 * 64 * 4);
  k_ball_query<<<cdiv(B * 64, 256), 256, 0, stream>>>(l0xyz, N0, nq1, 64, 0.04f, 64, gq1, B * 64);
  long Mq1 = (long)B * 64 * 64;                  // 65536
  f16* fq1 = (f16*)alloc(Mq1 * 160 * 2);
  k_group_feat<<<cdiv(Mq1, 256), 256, 0, stream>>>(l0xyz, nq1, gq1, l0p, 128, 128,
                                                   N0, 64, 64, 1, fq1, 160, (int)Mq1);
  f16* qa = (f16*)alloc(Mq1 * 128 * 2);
  f16* qb = (f16*)alloc(Mq1 * 128 * 2);
  gemm(fq1, 160, pn1L[0], 131, 64, qa, 64, Mq1, 1);
  gemm(qa, 64, pn1L[1], 64, 64, qb, 64, Mq1, 1);
  gemm(qb, 64, pn1L[2], 64, 128, qa, 128, Mq1, 1);
  f16* pb1 = (f16*)alloc((size_t)B * 64 * 128 * 2);
  k_maxpool<<<cdiv(B * 64 * 128, 256), 256, 0, stream>>>(qa, 128, 64, pb1, B * 64 * 128);

  // ================= pn2: SA(npoint=16, r=.4, K=64), feat 131 ==============
  int* idxq2 = (int*)alloc(B * 16 * 4);
  k_fps<<<B, 1024, 0, stream>>>(nq1, 64, 16, idxq2);
  float* nq2 = (float*)alloc(B * 16 * 3 * 4);
  k_gather_xyz<<<1, 256, 0, stream>>>(nq1, idxq2, nq2, 64, 16, B * 16);
  int* gq2 = (int*)alloc((size_t)B * 16 * 64 * 4);
  k_ball_query<<<1, 256, 0, stream>>>(nq1, 64, nq2, 16, 0.16f, 64, gq2, B * 16);
  long Mq2 = (long)B * 16 * 64;                  // 16384
  f16* fq2 = (f16*)alloc(Mq2 * 160 * 2);
  k_group_feat<<<cdiv(Mq2, 256), 256, 0, stream>>>(nq1, nq2, gq2, pb1, 128, 128,
                                                   64, 16, 64, 1, fq2, 160, (int)Mq2);
  f16* ra = (f16*)alloc(Mq2 * 256 * 2);
  f16* rb = (f16*)alloc(Mq2 * 256 * 2);
  gemm(fq2, 160, pn2L[0], 131, 128, ra, 128, Mq2, 1);
  gemm(ra, 128, pn2L[1], 128, 128, rb, 128, Mq2, 1);
  gemm(rb, 128, pn2L[2], 128, 256, ra, 256, Mq2, 1);
  f16* pb2 = (f16*)alloc((size_t)B * 16 * 256 * 2);
  k_maxpool<<<cdiv(B * 16 * 256, 256), 256, 0, stream>>>(ra, 256, 64, pb2, B * 16 * 256);

  // ================= pn3: group_all, feat 3+256=259 (pad 288) ==============
  long Mq3 = (long)B * 16;                       // 256
  f16* fq3 = (f16*)alloc(Mq3 * 288 * 2);
  k_groupall<<<1, 256, 0, stream>>>(nq2, pb2, 256, 256, fq3, 288, (int)Mq3);
  f16* sa = (f16*)alloc(Mq3 * 512 * 2);
  f16* sb = (f16*)alloc(Mq3 * 512 * 2);
  f16* sc = (f16*)alloc(Mq3 * 1024 * 2);
  gemm(fq3, 288, pn3L[0], 259, 256, sa, 256, Mq3, 1);
  gemm(sa, 256, pn3L[1], 256, 512, sb, 512, Mq3, 1);
  gemm(sb, 512, pn3L[2], 512, 1024, sc, 1024, Mq3, 1);
  f16* xg = (f16*)alloc((size_t)B * 1024 * 2);
  k_maxpool<<<cdiv(B * 1024, 256), 256, 0, stream>>>(sc, 1024, 16, xg, B * 1024);

  // ================= FC heads =================
  f16* h1 = (f16*)alloc(16 * 512 * 2);
  f16* h2 = (f16*)alloc(16 * 256 * 2);
  f16* lg = (f16*)alloc(16 * 32 * 2);
  gemm(xg, 1024, LP{fc1W, fc1b, bn1g, bn1e}, 1024, 512, h1, 512, 16, 1);
  gemm(h1, 512, LP{fc2W, fc2b, bn2g, bn2e}, 512, 256, h2, 256, 16, 1);
  gemm(h2, 256, LP{fc3W, fc3b, 0, 0}, 256, 10, lg, 32, 16, 0);
  f16* r4 = (f16*)alloc(16 * 512 * 2);
  f16* r5 = (f16*)alloc(16 * 256 * 2);
  f16* r6 = (f16*)alloc(16 * 32 * 2);
  gemm(xg, 1024, LP{fc4W, fc4b, 0, 0}, 1024, 512, r4, 512, 16, 1);
  gemm(r4, 512, LP{fc5W, fc5b, 0, 0}, 512, 256, r5, 256, 16, 1);
  gemm(r5, 256, LP{fc6W, fc6b, 0, 0}, 256, 7, r6, 32, 16, 1);

  k_head<<<1, 32, 0, stream>>>(lg, 32, 10, r6, 32, 7, (float*)d_out, B);
}
